// PARALLELNEURONLAYER_42202348650774
// MI455X (gfx1250) — matrix operand order repaired and checked by reading, not compiled, not run on hardware
//
#include <hip/hip_runtime.h>
#include <hip/hip_bf16.h>
#include <math.h>

// ---------------------------------------------------------------------------
// Problem constants (from reference):  T=8192, D=128, N=4096, NW=6
// windows [1,2,4,8,16,32]; window 32 = tiny attention path.
// Dominant cost: streaming comboW (24576x4096 fp32 = 402MB) once -> ~17us at
// 23.3 TB/s HBM. Matrix ops use v_wmma_f32_16x16x32_f16; the comboW stream
// uses b128 non-temporal loads + global_prefetch.
// ---------------------------------------------------------------------------

typedef __attribute__((ext_vector_type(16))) _Float16 v16h;
typedef __attribute__((ext_vector_type(8)))  float    v8f;
typedef __attribute__((ext_vector_type(4)))  float    v4f;

union FragH { v16h v; _Float16 h[16]; };
union FragF { v8f  v; float    f[8];  };

// 16-bit A-matrix 16x32 element->K map (per CDNA5 ISA):
//   e<8 : K = hi*8 + e        (hi = lane>=16)
//   e>=8: K = 8 + hi*8 + e
__device__ __forceinline__ int a_k(int e, int hi) {
  return (e < 8) ? (hi * 8 + e) : (8 + hi * 8 + e);
}
// 16-bit B-matrix 32x16 element->K map: same as A (e<8: hi*8+e; e>=8: 8+hi*8+e)
__device__ __forceinline__ int b_k(int e, int hi) { return a_k(e, hi); }

__device__ __forceinline__ v8f wmma16x16x32(const v16h a, const v16h b, const v8f c) {
  return __builtin_amdgcn_wmma_f32_16x16x32_f16(false, a, false, b, (short)0, c, false, false);
}

// ---------------------------------------------------------------------------
// Kernel 1: acts for last 16 tokens via WMMA, tanh+clip, suffix-window means
// for w = 1,2,4,8,16, scaled by norm_w[i], into weighted[i*4096 + n].
// One wave = one 16x16 output tile. 32 blocks x 8 waves = 256 tiles = 4096 cols.
// ---------------------------------------------------------------------------
__global__ void acts_windows_kernel(const float* __restrict__ X,    // [8192,128]
                                    const float* __restrict__ W,    // [128,4096]
                                    const float* __restrict__ B,    // [4096]
                                    const float* __restrict__ ww,   // [6]
                                    float* __restrict__ weighted)   // [6*4096]
{
  __shared__ float tile[8][16][16];
  const int tid  = threadIdx.x;
  const int wave = tid >> 5;
  const int lane = tid & 31;
  const int hi   = lane >> 4;
  const int m    = lane & 15;
  const int n0   = (blockIdx.x * 8 + wave) * 16;
  const int ncol = n0 + (lane & 15);

  const float* Xr = X + (size_t)(8192 - 16) * 128;   // last 16 token rows

  FragF c;
  #pragma unroll
  for (int r = 0; r < 8; ++r) c.f[r] = 0.0f;

  #pragma unroll
  for (int kk = 0; kk < 4; ++kk) {
    const int kb = kk * 32;
    FragH a, b;
    #pragma unroll
    for (int e = 0; e < 16; ++e) {
      a.h[e] = (_Float16)Xr[m * 128 + kb + a_k(e, hi)];
      b.h[e] = (_Float16)W[(size_t)(kb + b_k(e, hi)) * 4096 + ncol];
    }
    c.v = wmma16x16x32(a.v, b.v, c.v);
  }

  const float bias = B[ncol];
  #pragma unroll
  for (int r = 0; r < 8; ++r) {
    float v = tanhf(c.f[r] + bias);
    v = fminf(5.0f, fmaxf(-5.0f, v));
    tile[wave][hi * 8 + r][lane & 15] = v;   // row M = r + hi*8
  }
  __syncthreads();

  if (lane < 16) {
    float s6 = 0.0f;
    #pragma unroll
    for (int i = 0; i < 6; ++i) s6 += ww[i];
    const float inv = 1.0f / (s6 + 0.1f);

    const int wlen[5] = {1, 2, 4, 8, 16};
    float s = 0.0f;
    int wi = 0;
    #pragma unroll
    for (int cnt = 1; cnt <= 16; ++cnt) {
      s += tile[wave][16 - cnt][lane];       // suffix over last `cnt` tokens
      if (wi < 5 && cnt == wlen[wi]) {
        const float nw = (ww[wi] + 0.1f) * inv;
        weighted[wi * 4096 + n0 + lane] = nw * s / (float)cnt;
        ++wi;
      }
    }
  }
}

// ---------------------------------------------------------------------------
// Kernel 2: tiny attention over last 32 tokens, all GEMMs via WMMA.
// Single block, 256 threads (8 waves). Produces y[128] = LayerNorm(attn_mean).
// ---------------------------------------------------------------------------
__global__ void attn_kernel(const float* __restrict__ X,
                            const float* __restrict__ Wq, const float* __restrict__ bq,
                            const float* __restrict__ Wk, const float* __restrict__ bk,
                            const float* __restrict__ Wv, const float* __restrict__ bv,
                            const float* __restrict__ g,  const float* __restrict__ be,
                            float* __restrict__ y)        // [128] out
{
  __shared__ _Float16 Xh[32][128];
  __shared__ _Float16 Qh[32][128];
  __shared__ _Float16 Kh[32][128];
  __shared__ _Float16 Vh[32][128];
  __shared__ float    Sc[32][32];
  __shared__ float    At[32][128];
  __shared__ float    amean[128];
  __shared__ float    red[2];

  const int tid  = threadIdx.x;
  const int wave = tid >> 5;
  const int lane = tid & 31;
  const int hi   = lane >> 4;
  const int m    = lane & 15;
  const int nl   = lane & 15;

  // stage last-32-token rows as f16
  for (int i = tid; i < 32 * 128; i += 256) {
    const int t = i >> 7, d = i & 127;
    Xh[t][d] = (_Float16)X[(size_t)(8192 - 32 + t) * 128 + d];
  }
  __syncthreads();

  // --- Q/K/V projections: 3 * (2 M-tiles x 8 N-tiles), K = 4x32 ------------
  for (int t = wave; t < 48; t += 8) {          // uniform per wave
    const int which = t / 16;
    const int rem = t % 16;
    const int Mt = rem >> 3, Nt = rem & 7;

    const float* Wsel;
    const float* bsel;
    if (which == 0)      { Wsel = Wq; bsel = bq; }
    else if (which == 1) { Wsel = Wk; bsel = bk; }
    else                 { Wsel = Wv; bsel = bv; }

    FragF c;
    #pragma unroll
    for (int r = 0; r < 8; ++r) c.f[r] = 0.0f;
    #pragma unroll
    for (int kk = 0; kk < 4; ++kk) {
      const int kb = kk * 32;
      FragH a, b;
      #pragma unroll
      for (int e = 0; e < 16; ++e) {
        a.h[e] = Xh[Mt * 16 + m][kb + a_k(e, hi)];
        b.h[e] = (_Float16)Wsel[(size_t)(kb + b_k(e, hi)) * 128 + Nt * 16 + nl];
      }
      c.v = wmma16x16x32(a.v, b.v, c.v);
    }
    const float bias = bsel[Nt * 16 + nl];
    #pragma unroll
    for (int r = 0; r < 8; ++r) {
      const _Float16 v = (_Float16)(c.f[r] + bias);
      const int rr = Mt * 16 + hi * 8 + r;
      const int cc = Nt * 16 + nl;
      if (which == 0)      Qh[rr][cc] = v;
      else if (which == 1) Kh[rr][cc] = v;
      else                 Vh[rr][cc] = v;
    }
  }
  __syncthreads();

  // --- scores = Q @ K^T / sqrt(128): 2x2 tiles, waves 0..3 -----------------
  if (wave < 4) {
    const int Mt = wave >> 1, Nt = wave & 1;
    FragF c;
    #pragma unroll
    for (int r = 0; r < 8; ++r) c.f[r] = 0.0f;
    #pragma unroll
    for (int kk = 0; kk < 4; ++kk) {
      const int kb = kk * 32;
      FragH a, b;
      #pragma unroll
      for (int e = 0; e < 16; ++e) {
        a.h[e] = Qh[Mt * 16 + m][kb + a_k(e, hi)];
        b.h[e] = Kh[Nt * 16 + nl][kb + b_k(e, hi)];   // K^T[k][n] = K[n][k]
      }
      c.v = wmma16x16x32(a.v, b.v, c.v);
    }
    #pragma unroll
    for (int r = 0; r < 8; ++r)
      Sc[Mt * 16 + hi * 8 + r][Nt * 16 + nl] = c.f[r] * 0.08838834764831845f;
  }
  __syncthreads();

  // --- softmax per row (32 rows, one lane each) ----------------------------
  if (tid < 32) {
    float mx = -3.4e38f;
    for (int j = 0; j < 32; ++j) mx = fmaxf(mx, Sc[tid][j]);
    float s = 0.0f;
    for (int j = 0; j < 32; ++j) { const float e = __expf(Sc[tid][j] - mx); Sc[tid][j] = e; s += e; }
    const float is = 1.0f / s;
    for (int j = 0; j < 32; ++j) Sc[tid][j] *= is;
  }
  __syncthreads();

  // --- attended = P @ V : 2x8 tiles, K = 32 (single WMMA) ------------------
  for (int t = wave; t < 16; t += 8) {
    const int Mt = t >> 3, Nt = t & 7;
    FragF c;
    #pragma unroll
    for (int r = 0; r < 8; ++r) c.f[r] = 0.0f;
    FragH a, b;
    #pragma unroll
    for (int e = 0; e < 16; ++e) {
      a.h[e] = (_Float16)Sc[Mt * 16 + m][a_k(e, hi)];
      b.h[e] = Vh[b_k(e, hi)][Nt * 16 + nl];
    }
    c.v = wmma16x16x32(a.v, b.v, c.v);
    #pragma unroll
    for (int r = 0; r < 8; ++r)
      At[Mt * 16 + hi * 8 + r][Nt * 16 + nl] = c.f[r];
  }
  __syncthreads();

  // --- mean over 32 tokens -------------------------------------------------
  if (tid < 128) {
    float s = 0.0f;
    for (int t = 0; t < 32; ++t) s += At[t][tid];
    amean[tid] = s * (1.0f / 32.0f);
  }
  __syncthreads();

  // --- layernorm stats (serial over 128 elems — trivially cheap) -----------
  if (tid == 0) {
    float mu = 0.0f;
    for (int d = 0; d < 128; ++d) mu += amean[d];
    mu *= (1.0f / 128.0f);
    float var = 0.0f;
    for (int d = 0; d < 128; ++d) { const float t = amean[d] - mu; var += t * t; }
    var *= (1.0f / 128.0f);
    red[0] = mu;
    red[1] = rsqrtf(var + 1e-5f);
  }
  __syncthreads();

  if (tid < 128)
    y[tid] = (amean[tid] - red[0]) * red[1] * g[tid] + be[tid];
}

// ---------------------------------------------------------------------------
// Kernel 3: proj matvec: weighted[5*4096+n] = norm_w5 * (y @ projW + projb)[n]
// ---------------------------------------------------------------------------
__global__ void proj_kernel(const float* __restrict__ y,
                            const float* __restrict__ pW,   // [128,4096]
                            const float* __restrict__ pb,   // [4096]
                            const float* __restrict__ ww,   // [6]
                            float* __restrict__ weighted)
{
  __shared__ float ys[128];
  const int tid = threadIdx.x;
  if (tid < 128) ys[tid] = y[tid];
  __syncthreads();

  float s6 = 0.0f;
  #pragma unroll
  for (int i = 0; i < 6; ++i) s6 += ww[i];
  const float nw5 = (ww[5] + 0.1f) / (s6 + 0.1f);

  const int n = blockIdx.x * 256 + tid;
  float acc = pb[n];
  #pragma unroll 4
  for (int d = 0; d < 128; ++d)
    acc += ys[d] * pW[(size_t)d * 4096 + n];
  weighted[5 * 4096 + n] = acc * nw5;
}

// ---------------------------------------------------------------------------
// Kernel 4: comboW streaming matvec, b128 non-temporal loads.
// Grid: 4 column-tiles (1024 cols, 4/thread) x 192 K-chunks (128 rows) = 768
// blocks. Each wave load = 512B of comboW -> VMEM issue rate stays far below
// the 23.3 TB/s memory rate. Deterministic: per-chunk partials to ws.
// ---------------------------------------------------------------------------
__global__ void combo_partial_kernel(const float* __restrict__ weighted, // [24576]
                                     const float* __restrict__ cW,       // [24576,4096]
                                     float* __restrict__ partial)        // [192,4096]
{
  __shared__ float wk[128];
  const int tid = threadIdx.x;
  const int nt  = blockIdx.x & 3;       // 4 tiles of 1024 cols
  const int kc  = blockIdx.x >> 2;      // 192 chunks of 128 rows
  const int k0  = kc * 128;
  const int n0  = nt * 1024;

  if (tid < 128) wk[tid] = weighted[k0 + tid];
  __syncthreads();

  const v4f* base = (const v4f*)(cW + (size_t)k0 * 4096 + n0);  // row stride 1024 v4f
  v4f acc = {0.0f, 0.0f, 0.0f, 0.0f};
  for (int k = 0; k < 128; ++k) {
    const v4f* row = base + (size_t)k * 1024;
    if (k + 8 < 128)
      __builtin_prefetch((const float*)(row + (size_t)8 * 1024 + tid), 0, 0);
    const v4f v = __builtin_nontemporal_load(row + tid);
    acc += wk[k] * v;
  }
  v4f* dst = (v4f*)(partial + (size_t)kc * 4096 + n0);
  dst[tid] = acc;
}

__global__ void combo_reduce_kernel(const float* __restrict__ partial,  // [192,4096]
                                    const float* __restrict__ cb,       // [4096]
                                    float* __restrict__ out)            // [4096]
{
  const int i = blockIdx.x * 256 + threadIdx.x;     // v4 index, 1024 total
  v4f s = ((const v4f*)cb)[i];
  #pragma unroll 8
  for (int c = 0; c < 192; ++c)
    s += ((const v4f*)(partial + (size_t)c * 4096))[i];
  ((v4f*)out)[i] = s;
}

// Fallback path (tiny ws): init with bias then atomic-accumulate.
__global__ void init_out_kernel(const float* __restrict__ cb, float* __restrict__ out)
{
  const int n = blockIdx.x * 256 + threadIdx.x;
  out[n] = cb[n];
}

__global__ void combo_atomic_kernel(const float* __restrict__ weighted,
                                    const float* __restrict__ cW,
                                    float* __restrict__ out)
{
  __shared__ float wk[128];
  const int tid = threadIdx.x;
  const int nt  = blockIdx.x & 3;
  const int kc  = blockIdx.x >> 2;
  const int k0  = kc * 128;
  const int n0  = nt * 1024;
  if (tid < 128) wk[tid] = weighted[k0 + tid];
  __syncthreads();
  const v4f* base = (const v4f*)(cW + (size_t)k0 * 4096 + n0);
  v4f acc = {0.0f, 0.0f, 0.0f, 0.0f};
  for (int k = 0; k < 128; ++k) {
    const v4f* row = base + (size_t)k * 1024;
    const v4f v = __builtin_nontemporal_load(row + tid);
    acc += wk[k] * v;
  }
  float* dst = out + n0 + 4 * tid;
  atomicAdd(dst + 0, acc.x);
  atomicAdd(dst + 1, acc.y);
  atomicAdd(dst + 2, acc.z);
  atomicAdd(dst + 3, acc.w);
}

// ---------------------------------------------------------------------------
extern "C" void kernel_launch(void* const* d_in, const int* in_sizes, int n_in,
                              void* d_out, int out_size, void* d_ws, size_t ws_size,
                              hipStream_t stream)
{
  (void)in_sizes; (void)n_in; (void)out_size;
  const float* X   = (const float*)d_in[0];
  const float* nW  = (const float*)d_in[1];
  const float* nB  = (const float*)d_in[2];
  const float* Wq  = (const float*)d_in[3];
  const float* bq  = (const float*)d_in[4];
  const float* Wk  = (const float*)d_in[5];
  const float* bk  = (const float*)d_in[6];
  const float* Wv  = (const float*)d_in[7];
  const float* bv  = (const float*)d_in[8];
  const float* lng = (const float*)d_in[9];
  const float* lnb = (const float*)d_in[10];
  const float* pW  = (const float*)d_in[11];
  const float* pb  = (const float*)d_in[12];
  const float* ww  = (const float*)d_in[13];
  const float* cW  = (const float*)d_in[14];
  const float* cb  = (const float*)d_in[15];
  float* out = (float*)d_out;

  float* wsF      = (float*)d_ws;
  float* weighted = wsF;                  // 24576 floats
  float* y        = wsF + 24576;          // 128 floats
  float* partial  = wsF + 24576 + 128;    // 192*4096 floats (16B-aligned offset)

  const size_t need = (size_t)(24576 + 128 + 192 * 4096) * sizeof(float);

  acts_windows_kernel<<<dim3(32), dim3(256), 0, stream>>>(X, nW, nB, ww, weighted);
  attn_kernel<<<dim3(1), dim3(256), 0, stream>>>(X, Wq, bq, Wk, bk, Wv, bv, lng, lnb, y);
  proj_kernel<<<dim3(16), dim3(256), 0, stream>>>(y, pW, pb, ww, weighted);

  if (ws_size >= need) {
    combo_partial_kernel<<<dim3(768), dim3(256), 0, stream>>>(weighted, cW, partial);
    combo_reduce_kernel<<<dim3(4), dim3(256), 0, stream>>>(partial, cb, out);
  } else {
    init_out_kernel<<<dim3(16), dim3(256), 0, stream>>>(cb, out);
    combo_atomic_kernel<<<dim3(768), dim3(256), 0, stream>>>(weighted, cW, out);
  }
}
